// GNN_72155450573154
// MI455X (gfx1250) — matrix operand order repaired and checked by reading, not compiled, not run on hardware
//
#include <hip/hip_runtime.h>
#include <hip/hip_bf16.h>
#include <math.h>

typedef __attribute__((ext_vector_type(16))) _Float16 v16h;
typedef __attribute__((ext_vector_type(8)))  _Float16 v8h;
typedef __attribute__((ext_vector_type(8)))  float    v8f;

#define GN_N     50000
#define GN_E     800000
#define GN_ETOT  (GN_E + GN_N)      // edges + self loops
#define GN_D     128                // H*C
#define GN_H     4
#define GN_C     32
#define GN_ED    8
#define GN_NREL  16
#define NEG_SLOPE 0.2f

// ---------------------------------------------------------------- utilities

__device__ __forceinline__ void atomicMaxFloat(float* addr, float val) {
    // signed-int / unsigned-int split trick; buffer initialized to -inf
    if (val >= 0.0f) {
        atomicMax((int*)addr, __float_as_int(val));
    } else {
        atomicMin((unsigned int*)addr, __float_as_uint(val));
    }
}

// ---------------------------------------------------------------- prep

// colsum[j] = sum_k W_l0[k,j];  mean8[d] = mean of edge_emb over relations;
// WT1/WT2/WTm[j,k] = f16 transposed weights (WTm zero-padded from [128,100]).
__global__ void gnn_prep_kernel(const float* __restrict__ W_l0,
                                const float* __restrict__ W_l1,
                                const float* __restrict__ W_l2,
                                const float* __restrict__ mlp_W1,
                                const float* __restrict__ edge_emb,
                                float* __restrict__ colsum,
                                float* __restrict__ mean8,
                                _Float16* __restrict__ WT1,
                                _Float16* __restrict__ WT2,
                                _Float16* __restrict__ WTm) {
    int j = threadIdx.x;             // 0..127  (output column)
    float s = 0.0f;
    for (int k = 0; k < GN_C; ++k) s += W_l0[k * GN_D + j];
    colsum[j] = s;
    if (j < GN_ED) {
        float m = 0.0f;
        for (int r = 0; r < GN_NREL; ++r) m += edge_emb[r * GN_ED + j];
        mean8[j] = m * (1.0f / GN_NREL);
    }
    for (int k = 0; k < GN_D; ++k) {
        WT1[j * GN_D + k] = (_Float16)W_l1[k * GN_D + j];
        WT2[j * GN_D + k] = (_Float16)W_l2[k * GN_D + j];
        WTm[j * GN_D + k] = (j < 100) ? (_Float16)mlp_W1[k * 100 + j]
                                      : (_Float16)0.0f;
    }
}

// layer-0 "GEMM": h[n,j] = x[n] * colsum[j]   (x broadcast -> rank-1)
__global__ void gnn_layer0_h(const float* __restrict__ x,
                             const float* __restrict__ colsum,
                             float* __restrict__ Hh) {
    int idx = blockIdx.x * blockDim.x + threadIdx.x;
    if (idx >= GN_N * GN_D) return;
    Hh[idx] = x[idx >> 7] * colsum[idx & 127];
}

// Medge[d,h] = sum_c W_edge[l, d, h*C+c] * a_edge[l, h, c]   (8x4)
__global__ void gnn_medge_kernel(const float* __restrict__ W_edge_l,
                                 const float* __restrict__ att_edge_l,
                                 float* __restrict__ Medge) {
    int t = threadIdx.x;             // 0..31
    int d = t >> 2, h = t & 3;
    float s = 0.0f;
    for (int c = 0; c < GN_C; ++c)
        s += W_edge_l[d * GN_D + h * GN_C + c] * att_edge_l[h * GN_C + c];
    Medge[d * GN_H + h] = s;
}

// ---------------------------------------------------------------- WMMA GEMM
// Out[N,128] = X16[N,128] @ WT16^T; f16 operands, f32 accumulate.
// X16 row-major f16, WT16 column-of-W-major f16 (row j = output column j).
// One wave owns a 16-row block; A fragments persist across 8 column tiles.
// All fragment loads are contiguous 16B -> global_load_b128.
__global__ void gnn_wmma_gemm128(const _Float16* __restrict__ X16,
                                 const _Float16* __restrict__ WT16,
                                 float* __restrict__ Out, int nrows) {
    const int wave = (blockIdx.x * blockDim.x + threadIdx.x) >> 5;
    const int lane = threadIdx.x & 31;
    const int mbase = wave * 16;
    if (mbase >= nrows) return;      // wave-uniform: EXEC stays all-ones

    const int half16 = lane >> 4;    // 0 or 1
    const int m      = mbase + (lane & 15);
    const int ksel   = half16 << 3;  // A: lanes16-31 shifted by 8 in K

    union Frag { v16h v; v8h h[2]; };

    Frag afrag[4];
#pragma unroll
    for (int kk = 0; kk < 4; ++kk) {
        const _Float16* p = X16 + m * GN_D + (kk << 5) + ksel;
        afrag[kk].h[0] = *(const v8h*)(p);        // K = kk*32 + ksel + 0..7
        afrag[kk].h[1] = *(const v8h*)(p + 16);   // K = kk*32 + ksel + 16..23
    }

    const int nlane = lane & 15;
    const int kadd  = half16 << 3;   // B: lanes16-31 shifted by 8 in K
#pragma unroll
    for (int nb = 0; nb < 8; ++nb) {
        const int n = (nb << 4) + nlane;
        v8f c = {};
#pragma unroll
        for (int kk = 0; kk < 4; ++kk) {
            Frag b;
            const _Float16* p = WT16 + n * GN_D + (kk << 5) + kadd;
            b.h[0] = *(const v8h*)(p);            // K = kk*32 + kadd + 0..7
            b.h[1] = *(const v8h*)(p + 16);       // K = kk*32 + kadd + 16..23
            c = __builtin_amdgcn_wmma_f32_16x16x32_f16(
                    false, afrag[kk].v, false, b.v, (short)0, c, false, false);
        }
        const int mrow = mbase + (half16 << 3);
#pragma unroll
        for (int v = 0; v < 8; ++v)
            Out[(mrow + v) * GN_D + n] = c[v];
    }
}

// ---------------------------------------------------------------- attention

// s_src[n,h], s_dst[n,h]
__global__ void gnn_scores_kernel(const float* __restrict__ Hh,
                                  const float* __restrict__ a_src_l,
                                  const float* __restrict__ a_dst_l,
                                  float* __restrict__ ssrc,
                                  float* __restrict__ sdst) {
    int idx = blockIdx.x * blockDim.x + threadIdx.x;
    if (idx >= GN_N * GN_H) return;
    int n = idx >> 2, h = idx & 3;
    float s0 = 0.0f, s1 = 0.0f;
    const float4* row = (const float4*)(Hh + n * GN_D + h * GN_C);
    const float4* as  = (const float4*)(a_src_l + h * GN_C);
    const float4* ad  = (const float4*)(a_dst_l + h * GN_C);
#pragma unroll
    for (int c4 = 0; c4 < GN_C / 4; ++c4) {
        float4 v = row[c4], a = as[c4], b = ad[c4];
        s0 += v.x * a.x + v.y * a.y + v.z * a.z + v.w * a.w;
        s1 += v.x * b.x + v.y * b.y + v.z * b.z + v.w * b.w;
    }
    ssrc[idx] = s0;
    sdst[idx] = s1;
}

// zero accum, zero denom, set amax = -inf
__global__ void gnn_init_kernel(float* __restrict__ accum,
                                float* __restrict__ amax,
                                float* __restrict__ denom) {
    int idx = blockIdx.x * blockDim.x + threadIdx.x;
    if (idx >= GN_N * GN_D) return;
    accum[idx] = 0.0f;
    if (idx < GN_N * GN_H) {
        amax[idx]  = __int_as_float(0xFF800000); // -inf
        denom[idx] = 0.0f;
    }
}

// pass A: raw attention logits + leaky relu + segment max (atomic)
__global__ void gnn_passA_kernel(const int* __restrict__ ei,
                                 const int* __restrict__ etype,
                                 const float* __restrict__ edge_emb,
                                 const float* __restrict__ mean8,
                                 const float* __restrict__ Medge,
                                 const float* __restrict__ ssrc,
                                 const float* __restrict__ sdst,
                                 float* __restrict__ alphaws,
                                 float* __restrict__ amax) {
    int j = blockIdx.x * blockDim.x + threadIdx.x;
    if (j >= GN_ETOT) return;
    if (j + 4096 < GN_E) __builtin_prefetch(&ei[j + 4096], 0, 1);
    int s, d;
    float ea[GN_ED];
    if (j < GN_E) {
        s = ei[j];
        d = ei[GN_E + j];
        int t = etype[j];
#pragma unroll
        for (int e = 0; e < GN_ED; ++e) ea[e] = edge_emb[t * GN_ED + e];
    } else {
        s = d = j - GN_E;            // self loop, fill_value='mean'
#pragma unroll
        for (int e = 0; e < GN_ED; ++e) ea[e] = mean8[e];
    }
#pragma unroll
    for (int h = 0; h < GN_H; ++h) {
        float se = 0.0f;
#pragma unroll
        for (int e = 0; e < GN_ED; ++e) se += ea[e] * Medge[e * GN_H + h];
        float a = ssrc[s * GN_H + h] + sdst[d * GN_H + h] + se;
        a = (a > 0.0f) ? a : NEG_SLOPE * a;
        alphaws[j * GN_H + h] = a;
        atomicMaxFloat(&amax[d * GN_H + h], a);
    }
}

// pass B: exp(alpha - amax[dst]) + segment sum (atomic)
__global__ void gnn_passB_kernel(const int* __restrict__ ei,
                                 float* __restrict__ alphaws,
                                 const float* __restrict__ amax,
                                 float* __restrict__ denom) {
    int j = blockIdx.x * blockDim.x + threadIdx.x;
    if (j >= GN_ETOT) return;
    int d = (j < GN_E) ? ei[GN_E + j] : (j - GN_E);
#pragma unroll
    for (int h = 0; h < GN_H; ++h) {
        float e = expf(alphaws[j * GN_H + h] - amax[d * GN_H + h]);
        alphaws[j * GN_H + h] = e;
        atomicAdd(&denom[d * GN_H + h], e);
    }
}

// pass C: normalized weighted scatter of h[src] into accum[dst].
// One thread per (edge, 4-float chunk): b128 gather + 4 atomic f32 adds.
__global__ void gnn_passC_kernel(const int* __restrict__ ei,
                                 const float* __restrict__ alphaws,
                                 const float* __restrict__ denom,
                                 const float* __restrict__ Hh,
                                 float* __restrict__ accum) {
    long long idx = (long long)blockIdx.x * blockDim.x + threadIdx.x;
    if (idx >= (long long)GN_ETOT * (GN_D / 4)) return;
    int j  = (int)(idx >> 5);        // edge
    int q  = (int)(idx & 31);        // which float4 chunk (0..31)
    int h  = q >> 3;                 // 8 chunks per head
    int t0 = q << 2;                 // starting channel within row
    int s, d;
    if (j < GN_E) { s = ei[j]; d = ei[GN_E + j]; }
    else          { s = d = j - GN_E; }
    float w = alphaws[j * GN_H + h] / (denom[d * GN_H + h] + 1e-16f);
    float4 v = *(const float4*)(Hh + s * GN_D + t0);
    float* o = accum + d * GN_D + t0;
    atomicAdd(o + 0, v.x * w);
    atomicAdd(o + 1, v.y * w);
    atomicAdd(o + 2, v.z * w);
    atomicAdd(o + 3, v.w * w);
}

// bias + (residual) + relu -> next x (f32 for residual, f16 for WMMA A)
__global__ void gnn_finalize_kernel(const float* __restrict__ accum,
                                    const float* __restrict__ bias_l,
                                    const float* __restrict__ xin,
                                    float* __restrict__ xout,
                                    _Float16* __restrict__ x16out,
                                    int use_residual) {
    int idx = blockIdx.x * blockDim.x + threadIdx.x;
    if (idx >= GN_N * GN_D) return;
    float v = accum[idx] + bias_l[idx & 127];
    if (use_residual) v += xin[idx];
    v = (v > 0.0f) ? v : 0.0f;
    xout[idx]   = v;
    x16out[idx] = (_Float16)v;
}

// MLP tail: relu(h + b1) dot W2 + b2 -> sigmoid
__global__ void gnn_mlp_final(const float* __restrict__ Hh,
                              const float* __restrict__ b1,
                              const float* __restrict__ W2,
                              const float* __restrict__ b2,
                              float* __restrict__ out) {
    int n = blockIdx.x * blockDim.x + threadIdx.x;
    if (n >= GN_N) return;
    float acc = 0.0f;
    const float* row = Hh + n * GN_D;
    for (int j = 0; j < 100; ++j) {
        float v = row[j] + b1[j];
        v = (v > 0.0f) ? v : 0.0f;
        acc += v * W2[j];
    }
    acc += b2[0];
    out[n] = 1.0f / (1.0f + expf(-acc));
}

// ---------------------------------------------------------------- launch

extern "C" void kernel_launch(void* const* d_in, const int* in_sizes, int n_in,
                              void* d_out, int out_size, void* d_ws, size_t ws_size,
                              hipStream_t stream) {
    const float* x        = (const float*)d_in[0];
    const int*   ei       = (const int*)  d_in[1];   // [2,E]
    const int*   etype    = (const int*)  d_in[2];   // [E]
    const float* edge_emb = (const float*)d_in[3];   // [16,8]
    const float* W_l0     = (const float*)d_in[4];
    const float* W_l1     = (const float*)d_in[5];
    const float* W_l2     = (const float*)d_in[6];
    const float* att_src  = (const float*)d_in[7];   // [3,4,32]
    const float* att_dst  = (const float*)d_in[8];
    const float* att_edge = (const float*)d_in[9];
    const float* W_edge   = (const float*)d_in[10];  // [3,8,128]
    const float* bias     = (const float*)d_in[11];  // [3,128]
    const float* mlp_W1   = (const float*)d_in[12];  // [128,100]
    const float* mlp_b1   = (const float*)d_in[13];
    const float* mlp_W2   = (const float*)d_in[14];
    const float* mlp_b2   = (const float*)d_in[15];
    float* out = (float*)d_out;

    // workspace carve-up (256B aligned slabs)
    char* ws = (char*)d_ws;
    size_t o = 0;
    auto alloc = [&](size_t bytes) { char* p = ws + o; o += (bytes + 255) & ~(size_t)255; return (void*)p; };
    float* xA      = (float*)alloc((size_t)GN_N * GN_D * 4);
    float* xB      = (float*)alloc((size_t)GN_N * GN_D * 4);
    float* Hh      = (float*)alloc((size_t)GN_N * GN_D * 4);
    float* accum   = (float*)alloc((size_t)GN_N * GN_D * 4);
    _Float16* x16  = (_Float16*)alloc((size_t)GN_N * GN_D * 2);
    float* ssrc    = (float*)alloc((size_t)GN_N * GN_H * 4);
    float* sdst    = (float*)alloc((size_t)GN_N * GN_H * 4);
    float* amax    = (float*)alloc((size_t)GN_N * GN_H * 4);
    float* denom   = (float*)alloc((size_t)GN_N * GN_H * 4);
    float* alphaws = (float*)alloc((size_t)GN_ETOT * GN_H * 4);
    float* colsum  = (float*)alloc(GN_D * 4);
    float* mean8   = (float*)alloc(GN_ED * 4);
    float* Medge   = (float*)alloc(GN_ED * GN_H * 4);
    _Float16* WT1  = (_Float16*)alloc(GN_D * GN_D * 2);
    _Float16* WT2  = (_Float16*)alloc(GN_D * GN_D * 2);
    _Float16* WTm  = (_Float16*)alloc(GN_D * GN_D * 2);
    (void)ws_size; (void)n_in; (void)in_sizes; (void)out_size;

    const int T = 256;
    const int bND   = (GN_N * GN_D + T - 1) / T;          // 25000
    const int bNH   = (GN_N * GN_H + T - 1) / T;
    const int bE    = (GN_ETOT + T - 1) / T;
    const long long scat = (long long)GN_ETOT * (GN_D / 4);
    const int bSC   = (int)((scat + T - 1) / T);
    const int tiles = GN_N / 16;                          // 3125 (exact)
    const int bGEMM = (tiles + (T / 32) - 1) / (T / 32);  // 8 waves/block

    gnn_prep_kernel<<<1, 128, 0, stream>>>(W_l0, W_l1, W_l2, mlp_W1, edge_emb,
                                           colsum, mean8, WT1, WT2, WTm);

    const _Float16* WTl[3] = { nullptr, WT1, WT2 };
    float* xcur = nullptr;
    for (int l = 0; l < 3; ++l) {
        // h = x @ W  (layer 0 is rank-1; layers 1-2 use WMMA on f16 tiles)
        if (l == 0)
            gnn_layer0_h<<<bND, T, 0, stream>>>(x, colsum, Hh);
        else
            gnn_wmma_gemm128<<<bGEMM, T, 0, stream>>>(x16, WTl[l], Hh, GN_N);

        gnn_scores_kernel<<<bNH, T, 0, stream>>>(Hh, att_src + l * GN_D,
                                                 att_dst + l * GN_D, ssrc, sdst);
        gnn_medge_kernel<<<1, 32, 0, stream>>>(W_edge + l * GN_ED * GN_D,
                                               att_edge + l * GN_D, Medge);
        gnn_init_kernel<<<bND, T, 0, stream>>>(accum, amax, denom);
        gnn_passA_kernel<<<bE, T, 0, stream>>>(ei, etype, edge_emb, mean8, Medge,
                                               ssrc, sdst, alphaws, amax);
        gnn_passB_kernel<<<bE, T, 0, stream>>>(ei, alphaws, amax, denom);
        gnn_passC_kernel<<<bSC, T, 0, stream>>>(ei, alphaws, denom, Hh, accum);

        float* xnext = (l == 0) ? xA : ((l == 1) ? xB : xA);
        gnn_finalize_kernel<<<bND, T, 0, stream>>>(accum, bias + l * GN_D,
                                                   xcur ? xcur : accum, xnext,
                                                   x16, l > 0 ? 1 : 0);
        xcur = xnext;
    }

    // MLP: hid = x @ W1pad (WMMA f16), then fused relu/dot/sigmoid
    gnn_wmma_gemm128<<<bGEMM, T, 0, stream>>>(x16, WTm, Hh, GN_N);
    gnn_mlp_final<<<(GN_N + T - 1) / T, T, 0, stream>>>(Hh, mlp_b1, mlp_W2,
                                                        mlp_b2, out);
}
